// EventEncoder_82617990906318
// MI455X (gfx1250) — matrix operand order repaired and checked by reading, not compiled, not run on hardware
//
#include <hip/hip_runtime.h>
#include <hip/hip_bf16.h>
#include <math.h>

#define S 4096
#define NSEG 128
#define EVLEN 32
#define VOCAB 32002
#define D 1024
#define H 8
#define DH 128
#define DFF 4096
#define L 2

typedef unsigned short u16;
typedef __attribute__((ext_vector_type(16))) __bf16 v16bf;
typedef __attribute__((ext_vector_type(8)))  __bf16 v8bf;
typedef __attribute__((ext_vector_type(8)))  float  v8f;

__device__ __forceinline__ u16 f32_to_bf16(float f) {
    union { float f; unsigned int u; } x; x.f = f;
    unsigned int u = x.u;
    unsigned int r = (u + 0x7FFFu + ((u >> 16) & 1u)) >> 16;
    return (u16)r;
}

// ---------------------------------------------------------------------------
// Convert f32 weight [K,N] row-major -> bf16 swizzled B-fragment layout:
// Wsw[((kt*ntiles + nt)*32 + lane)*16 + i] = W[kt*32 + (lane>>4)*16 + i][nt*16 + (lane&15)]
// ---------------------------------------------------------------------------
__global__ void convert_swizzle_kernel(const float* __restrict__ W, u16* __restrict__ Wsw,
                                       int K, int N) {
    size_t idx = (size_t)blockIdx.x * 256 + threadIdx.x;
    size_t total = (size_t)K * N;
    if (idx >= total) return;
    int i    = (int)(idx & 15);
    int lane = (int)((idx >> 4) & 31);
    size_t tile = idx >> 9;
    int ntiles = N >> 4;
    int nt = (int)(tile % ntiles);
    int kt = (int)(tile / ntiles);
    int kk = kt * 32 + (lane >> 4) * 16 + i;
    int nn = nt * 16 + (lane & 15);
    Wsw[idx] = f32_to_bf16(W[(size_t)kk * N + nn]);
}

// ---------------------------------------------------------------------------
// Embedding gather: h[t, d] = emb[ids[t], d]
// ---------------------------------------------------------------------------
__global__ void embed_kernel(const int* __restrict__ ids, const float* __restrict__ emb,
                             float* __restrict__ h) {
    size_t idx = (size_t)blockIdx.x * 256 + threadIdx.x;   // S*D threads
    int t = (int)(idx >> 10);
    int d = (int)(idx & 1023);
    h[idx] = emb[(size_t)ids[t] * D + d];
}

// ---------------------------------------------------------------------------
// RMSNorm row -> bf16 output (row-major, feeds GEMM A-side)
// ---------------------------------------------------------------------------
__global__ void rmsnorm_bf16_kernel(const float* __restrict__ h, const float* __restrict__ g,
                                    u16* __restrict__ xb) {
    int t = blockIdx.x;
    __shared__ float red[256];
    float s = 0.f;
    for (int d = threadIdx.x; d < D; d += 256) {
        float v = h[(size_t)t * D + d];
        s += v * v;
    }
    red[threadIdx.x] = s;
    __syncthreads();
    for (int o = 128; o > 0; o >>= 1) {
        if (threadIdx.x < (unsigned)o) red[threadIdx.x] += red[threadIdx.x + o];
        __syncthreads();
    }
    float inv = rsqrtf(red[0] / (float)D + 1e-6f);
    for (int d = threadIdx.x; d < D; d += 256) {
        xb[(size_t)t * D + d] = f32_to_bf16(h[(size_t)t * D + d] * inv * g[d]);
    }
}

// ---------------------------------------------------------------------------
// WMMA GEMM: C[M,N] = A[M,K] (bf16 row-major) x Wsw[K,N] (bf16 swizzled)
// MODE 0: store f32     MODE 1: store f32 += residual     MODE 2: silu -> bf16
// block = 256 threads (8 waves, 2(M) x 4(N)); wave computes a 32x64 tile
// (8 accumulators = 64 VGPRs; 2 A frags reused 4x; each B frag reused 2x).
// Block tile 64x256. grid (M/64, N/256). ~130 VGPRs -> no spills.
// ---------------------------------------------------------------------------
template <int MODE>
__global__ void gemm_wmma_kernel(const u16* __restrict__ A, const u16* __restrict__ Wsw,
                                 int M, int N, int K,
                                 float* __restrict__ Cf, const float* __restrict__ Res,
                                 u16* __restrict__ Cb) {
    const int lane = threadIdx.x & 31;
    const int wave = threadIdx.x >> 5;          // 0..7
    const int hl   = lane >> 4;                 // 0/1
    const int r    = lane & 15;
    const int m0   = blockIdx.x * 64 + (wave & 1) * 32;
    const int n0   = blockIdx.y * 256 + (wave >> 1) * 64;
    const int ntiles = N >> 4;

    v8f acc[8] = {};

    for (int k0 = 0; k0 < K; k0 += 32) {
        // A fragments (contiguous k = 16*hl + s, matching swizzled B), 2 m-subtiles
        v16bf a[2];
#pragma unroll
        for (int i = 0; i < 2; ++i) {
            const u16* arow = A + (size_t)(m0 + i * 16 + r) * K + k0 + hl * 16;
            v8bf alo = *(const v8bf*)(arow);
            v8bf ahi = *(const v8bf*)(arow + 8);
            a[i] = __builtin_shufflevector(alo, ahi,
                        0, 1, 2, 3, 4, 5, 6, 7, 8, 9, 10, 11, 12, 13, 14, 15);
        }

        const int kt = k0 >> 5;
        const size_t btile0 = ((size_t)kt * ntiles + (n0 >> 4)) * 512 + (size_t)lane * 16;
#pragma unroll
        for (int j = 0; j < 4; ++j) {
            const u16* bp = Wsw + btile0 + (size_t)j * 512;
            v8bf blo = *(const v8bf*)(bp);
            v8bf bhi = *(const v8bf*)(bp + 8);
            v16bf b = __builtin_shufflevector(blo, bhi,
                        0, 1, 2, 3, 4, 5, 6, 7, 8, 9, 10, 11, 12, 13, 14, 15);
#pragma unroll
            for (int i = 0; i < 2; ++i) {
                acc[i * 4 + j] = __builtin_amdgcn_wmma_f32_16x16x32_bf16(
                    false, a[i], false, b, (short)0, acc[i * 4 + j], false, false);
            }
        }
    }

    // Epilogue: element v of acc[i*4+j] is (m0 + i*16 + v + hl*8, n0 + j*16 + r)
#pragma unroll
    for (int i = 0; i < 2; ++i) {
#pragma unroll
        for (int j = 0; j < 4; ++j) {
            int n = n0 + j * 16 + r;
#pragma unroll
            for (int v = 0; v < 8; ++v) {
                int m = m0 + i * 16 + v + hl * 8;
                size_t idx = (size_t)m * N + n;
                float c = acc[i * 4 + j][v];
                if constexpr (MODE == 0) {
                    Cf[idx] = c;
                } else if constexpr (MODE == 1) {
                    Cf[idx] = c + Res[idx];
                } else {
                    float sv = c / (1.f + __expf(-c));   // silu
                    Cb[idx] = f32_to_bf16(sv);
                }
            }
        }
    }
}

// ---------------------------------------------------------------------------
// RoPE applied in-place to q and k (f32, [S, H*DH])
// ---------------------------------------------------------------------------
__global__ void rope_kernel(float* __restrict__ q, float* __restrict__ k,
                            const int* __restrict__ pos) {
    int idx = blockIdx.x * 256 + threadIdx.x;     // S*H*(DH/2) = 2M
    int p  = idx & 63;
    int hh = (idx >> 6) & 7;
    int t  = idx >> 9;
    float inv_freq = __powf(10000.f, -(2.f * (float)p) / (float)DH);
    float ang = (float)pos[t] * inv_freq;
    float c = __cosf(ang), s = __sinf(ang);
    size_t base = (size_t)t * D + hh * DH + 2 * p;
    float q1 = q[base], q2 = q[base + 1];
    q[base]     = q1 * c - q2 * s;
    q[base + 1] = q1 * s + q2 * c;
    float k1 = k[base], k2 = k[base + 1];
    k[base]     = k1 * c - k2 * s;
    k[base + 1] = k1 * s + k2 * c;
}

// ---------------------------------------------------------------------------
// Per-segment causal attention. One wave per (segment, head); lane = query row.
// Writes bf16 context, feeding the Wo GEMM.
// ---------------------------------------------------------------------------
__global__ void attn_kernel(const float* __restrict__ q, const float* __restrict__ k,
                            const float* __restrict__ v, u16* __restrict__ ob) {
    int wid  = blockIdx.x * (blockDim.x >> 5) + (threadIdx.x >> 5);  // 0..1023
    int head = wid & 7;
    int seg  = wid >> 3;
    int lane = threadIdx.x & 31;
    int ti   = seg * EVLEN + lane;
    const float scale = 0.088388347648318447f;   // 1/sqrt(128)

    float p[EVLEN];
    float mx = -1e30f;
#pragma unroll 4
    for (int j = 0; j < EVLEN; ++j) {
        const float* qr = q + (size_t)ti * D + head * DH;
        const float* kr = k + (size_t)(seg * EVLEN + j) * D + head * DH;
        float s = 0.f;
#pragma unroll
        for (int d = 0; d < DH; ++d) s += qr[d] * kr[d];
        s *= scale;
        if (j > lane) s = -1e9f;                 // causal mask within segment
        p[j] = s;
        mx = fmaxf(mx, s);
    }
    float sum = 0.f;
#pragma unroll
    for (int j = 0; j < EVLEN; ++j) { p[j] = __expf(p[j] - mx); sum += p[j]; }
    float inv = 1.f / sum;
#pragma unroll
    for (int j = 0; j < EVLEN; ++j) p[j] *= inv;

    for (int d = 0; d < DH; ++d) {
        float a = 0.f;
#pragma unroll 8
        for (int j = 0; j < EVLEN; ++j)
            a += p[j] * v[(size_t)(seg * EVLEN + j) * D + head * DH + d];
        ob[(size_t)ti * D + head * DH + d] = f32_to_bf16(a);
    }
}

// ---------------------------------------------------------------------------
// Final RMSNorm + gather of last token of each segment -> d_out [128, 1024] f32
// ---------------------------------------------------------------------------
__global__ void final_gather_kernel(const float* __restrict__ h, const float* __restrict__ lnf,
                                    const int* __restrict__ seq_varlen, float* __restrict__ out) {
    int row = blockIdx.x;                        // 0..127
    int tok = -1;
    for (int i = 0; i <= row; ++i) tok += seq_varlen[i];
    __shared__ float red[256];
    float s = 0.f;
    for (int d = threadIdx.x; d < D; d += 256) {
        float vv = h[(size_t)tok * D + d];
        s += vv * vv;
    }
    red[threadIdx.x] = s;
    __syncthreads();
    for (int o = 128; o > 0; o >>= 1) {
        if (threadIdx.x < (unsigned)o) red[threadIdx.x] += red[threadIdx.x + o];
        __syncthreads();
    }
    float inv = rsqrtf(red[0] / (float)D + 1e-6f);
    for (int d = threadIdx.x; d < D; d += 256) {
        out[(size_t)row * D + d] = h[(size_t)tok * D + d] * inv * lnf[d];
    }
}

// ---------------------------------------------------------------------------
extern "C" void kernel_launch(void* const* d_in, const int* in_sizes, int n_in,
                              void* d_out, int out_size, void* d_ws, size_t ws_size,
                              hipStream_t stream) {
    const int*   input_ids = (const int*)d_in[0];
    const int*   pos_ids   = (const int*)d_in[1];
    const int*   seq_vl    = (const int*)d_in[2];
    const float* emb       = (const float*)d_in[3];
    const float* Wq        = (const float*)d_in[4];
    const float* Wk        = (const float*)d_in[5];
    const float* Wv        = (const float*)d_in[6];
    const float* Wo        = (const float*)d_in[7];
    const float* W1        = (const float*)d_in[8];
    const float* W2        = (const float*)d_in[9];
    const float* ln1       = (const float*)d_in[10];
    const float* ln2       = (const float*)d_in[11];
    const float* lnf       = (const float*)d_in[12];
    float*       out       = (float*)d_out;

    // ---- workspace layout ----
    const size_t DD = (size_t)D * D;          // 1,048,576
    const size_t DF = (size_t)D * DFF;        // 4,194,304
    const size_t perLayerW = 4 * DD + 2 * DF; // elems (bf16)
    char* base = (char*)d_ws;
    u16*   wsw = (u16*)base;                                  // 48 MB swizzled bf16 weights
    float* h   = (float*)(base + perLayerW * L * sizeof(u16));
    u16*   xb  = (u16*)((char*)h + (size_t)S * D * sizeof(float));
    float* q   = (float*)((char*)xb + (size_t)S * D * sizeof(u16));
    float* kk  = q  + (size_t)S * D;
    float* vv  = kk + (size_t)S * D;
    u16*   ob  = (u16*)(vv + (size_t)S * D);
    u16*   ffb = ob + (size_t)S * D;                          // S x DFF bf16

    // per-layer swizzled weight pointers
    u16* WqS[L]; u16* WkS[L]; u16* WvS[L]; u16* WoS[L]; u16* W1S[L]; u16* W2S[L];
    for (int l = 0; l < L; ++l) {
        u16* p = wsw + (size_t)l * perLayerW;
        WqS[l] = p;            WkS[l] = p + DD;       WvS[l] = p + 2 * DD;
        WoS[l] = p + 3 * DD;   W1S[l] = p + 4 * DD;   W2S[l] = p + 4 * DD + DF;
    }

    // ---- 1) convert + swizzle all weights to bf16 ----
    for (int l = 0; l < L; ++l) {
        int gDD = (int)((DD + 255) / 256), gDF = (int)((DF + 255) / 256);
        convert_swizzle_kernel<<<gDD, 256, 0, stream>>>(Wq + l * DD, WqS[l], D, D);
        convert_swizzle_kernel<<<gDD, 256, 0, stream>>>(Wk + l * DD, WkS[l], D, D);
        convert_swizzle_kernel<<<gDD, 256, 0, stream>>>(Wv + l * DD, WvS[l], D, D);
        convert_swizzle_kernel<<<gDD, 256, 0, stream>>>(Wo + l * DD, WoS[l], D, D);
        convert_swizzle_kernel<<<gDF, 256, 0, stream>>>(W1 + l * DF, W1S[l], D, DFF);
        convert_swizzle_kernel<<<gDF, 256, 0, stream>>>(W2 + l * DF, W2S[l], DFF, D);
    }

    // ---- 2) embedding gather ----
    embed_kernel<<<(S * D) / 256, 256, 0, stream>>>(input_ids, emb, h);

    // ---- 3) transformer layers ----
    dim3 gProj(S / 64, D / 256);      // N = 1024 -> (64, 4)
    dim3 gFF1(S / 64, DFF / 256);     // N = 4096 -> (64, 16)
    for (int l = 0; l < L; ++l) {
        rmsnorm_bf16_kernel<<<S, 256, 0, stream>>>(h, ln1 + (size_t)l * D, xb);
        gemm_wmma_kernel<0><<<gProj, 256, 0, stream>>>(xb, WqS[l], S, D, D, q,  nullptr, nullptr);
        gemm_wmma_kernel<0><<<gProj, 256, 0, stream>>>(xb, WkS[l], S, D, D, kk, nullptr, nullptr);
        gemm_wmma_kernel<0><<<gProj, 256, 0, stream>>>(xb, WvS[l], S, D, D, vv, nullptr, nullptr);
        rope_kernel<<<(S * H * (DH / 2)) / 256, 256, 0, stream>>>(q, kk, pos_ids);
        attn_kernel<<<(NSEG * H) / 8, 256, 0, stream>>>(q, kk, vv, ob);
        gemm_wmma_kernel<1><<<gProj, 256, 0, stream>>>(ob, WoS[l], S, D, D, h, h, nullptr);
        rmsnorm_bf16_kernel<<<S, 256, 0, stream>>>(h, ln2 + (size_t)l * D, xb);
        gemm_wmma_kernel<2><<<gFF1, 256, 0, stream>>>(xb, W1S[l], S, DFF, D, nullptr, nullptr, ffb);
        gemm_wmma_kernel<1><<<gProj, 256, 0, stream>>>(ffb, W2S[l], S, D, DFF, h, h, nullptr);
    }

    // ---- 4) final norm + gather ----
    final_gather_kernel<<<NSEG, 256, 0, stream>>>(h, lnf, seq_vl, out);
}
